// CRF_25125558682279
// MI455X (gfx1250) — matrix operand order repaired and checked by reading, not compiled, not run on hardware
//
#include <hip/hip_runtime.h>
#include <hip/hip_bf16.h>

// CRF loss on gfx1250: forward algorithm recast as exp-domain GEMM so the
// K x K logsumexp contraction runs on v_wmma_f32_16x16x32_f16.
//
//   E[i][j]        = exp(trans[i][j])                (constant, f16, LDS)
//   p[b][j]        = exp(alpha[b][j] - rowmax[b])    (f16, LDS, per step)
//   y[b][i]        = sum_j p[b][j] * E[i][j]         (WMMA, f32 accum)
//   alpha'[b][i]   = log(y) + rowmax[b] + feat[b][t][i]
//
// 32 workgroups x 256 threads; each WG owns 16 batch rows, each of its 8
// waves owns a 16-wide N block and keeps its 4 B-fragments (E) in VGPRs for
// the whole T loop. Feat tiles are double-buffered in LDS via
// global_load_async_to_lds_b128 (ASYNCcnt), hiding HBM latency behind the
// WMMA/log phases of the sequential recurrence. exp/log use the raw
// v_exp_f32/v_log_f32 ops (base-2 + constant scale) to keep the critical
// chain short.

#define CRF_B 512
#define CRF_T 512
#define CRF_K 128
#define CRF_START 126
#define CRF_STOP 127
#define CRF_NEG -10000.0f

#define LOG2E 1.44269504088896340736f
#define LN2   0.69314718055994530942f

typedef _Float16 v16h __attribute__((ext_vector_type(16)));
typedef _Float16 v8h  __attribute__((ext_vector_type(8)));
typedef float    v8f  __attribute__((ext_vector_type(8)));

__global__ __launch_bounds__(256) void crf_forward_kernel(
    const float* __restrict__ feats,      // [B][T][K]
    const float* __restrict__ trans,      // [K][K]
    float* __restrict__ out)              // scalar accumulator
{
    constexpr int K  = CRF_K;
    constexpr int T  = CRF_T;
    constexpr int ES = 136;   // E row stride in f16 (128 + 8 pad: bank spread)
    constexpr int PS = 136;   // p row stride in f16
    constexpr int AS = 129;   // alpha row stride in f32
    constexpr int FBUF = 16 * K;              // floats per feat buffer
    constexpr unsigned FBYTES = FBUF * 4;     // 8192 B per buffer

    __shared__ _Float16 E_lds[K * ES];        // E[i][j] = exp(trans[i][j])
    __shared__ _Float16 p_lds[16 * PS];       // exp(alpha - rowmax), f16
    __shared__ float    alpha_lds[16 * AS];   // running alpha, f32
    __shared__ float    feat_lds[2 * FBUF];   // double-buffered feat tile
    __shared__ float    pmax[16][16];
    __shared__ float    rowmax[16];

    const int tid  = threadIdx.x;
    const int wave = tid >> 5;        // 0..7
    const int lane = tid & 31;
    const int half = lane >> 4;       // 0 or 1
    const int l15  = lane & 15;
    const int nb   = wave * 16;       // this wave's N block (dest-tag columns)
    const int b0   = blockIdx.x * 16; // this WG's batch rows

    // ---- async feat staging: thread tid owns floats [8*tid, 8*tid+8) ------
    // of the 16x128 tile = two b128 chunks (offset:16 covers the second;
    // IOFFSET adds to both the global and LDS addresses).
    const int frow = tid >> 4;              // tile row 0..15
    const int fcol = (tid & 15) * 8;        // starting column (multiple of 8)
    const unsigned lds_feat_base =
        (unsigned)(size_t)&feat_lds[frow * K + fcol];
    unsigned long long gfeat =
        (unsigned long long)(size_t)(feats + ((size_t)(b0 + frow) * T) * K + fcol);

    // issue t=0 tile into buffer 0
    __asm__ volatile("global_load_async_to_lds_b128 %0, %1, off"
                     :: "v"(lds_feat_base), "v"(gfeat) : "memory");
    __asm__ volatile("global_load_async_to_lds_b128 %0, %1, off offset:16"
                     :: "v"(lds_feat_base), "v"(gfeat) : "memory");
    gfeat += (unsigned long long)K * 4;     // advance one timestep (512 B)

    // ---- one-time: E = exp(trans) into LDS as f16 -------------------------
    for (int idx = tid; idx < K * K; idx += 256) {
        int r = idx >> 7, cc = idx & (K - 1);
        E_lds[r * ES + cc] = (_Float16)__builtin_amdgcn_exp2f(trans[idx] * LOG2E);
    }
    // ---- one-time: alpha init: NEG everywhere, 0 at START -----------------
    for (int idx = tid; idx < 16 * K; idx += 256) {
        int r = idx >> 7, cc = idx & (K - 1);
        alpha_lds[r * AS + cc] = (cc == CRF_START) ? 0.0f : CRF_NEG;
    }
    __syncthreads();

    // ---- preload loop-invariant B fragments (E is constant over t) --------
    // B[k][n] = E[n][k]; lane n<16 holds col n, K=k0..k0+7 then k0+16..k0+23;
    // lane n+16 holds K=k0+8..k0+15 then k0+24..k0+31. Row of E is K-contiguous.
    v16h bf[4];
#pragma unroll
    for (int kk = 0; kk < 4; ++kk) {
        const int kb = kk * 32 + half * 8;
        v8h blo = *(const v8h*)&E_lds[(nb + l15) * ES + kb];
        v8h bhi = *(const v8h*)&E_lds[(nb + l15) * ES + kb + 16];
        bf[kk] = __builtin_shufflevector(blo, bhi,
            0, 1, 2, 3, 4, 5, 6, 7, 8, 9, 10, 11, 12, 13, 14, 15);
    }

    const int prow = tid >> 4;          // phase-1/2 row (0..15)
    const int pcol = (tid & 15) * 8;    // 8 contiguous columns per thread

    for (int t = 0; t < T; ++t) {
        // -- phase 1: per-row max of alpha (stability shift) ----------------
        float mloc = alpha_lds[prow * AS + pcol];
#pragma unroll
        for (int i = 1; i < 8; ++i)
            mloc = fmaxf(mloc, alpha_lds[prow * AS + pcol + i]);
        pmax[prow][tid & 15] = mloc;
        __syncthreads();
        if (tid < 16) {
            float mm = pmax[tid][0];
#pragma unroll
            for (int j = 1; j < 16; ++j) mm = fmaxf(mm, pmax[tid][j]);
            rowmax[tid] = mm;
        }
        __syncthreads();

        // -- phase 2: p = exp2((alpha - rowmax)*log2e) -> f16 LDS -----------
        {
            const float rm = rowmax[prow];
#pragma unroll
            for (int i = 0; i < 8; ++i) {
                float a = alpha_lds[prow * AS + pcol + i];
                p_lds[prow * PS + pcol + i] =
                    (_Float16)__builtin_amdgcn_exp2f((a - rm) * LOG2E);
            }
        }
        // fence this step's async feat tile, then make it WG-visible
        __asm__ volatile("s_wait_asynccnt 0" ::: "memory");
        __syncthreads();

        // -- kick off async load of the t+1 feat tile into the other buffer -
        if (t + 1 < T) {
            const unsigned ldst = lds_feat_base + (((t + 1) & 1) ? FBYTES : 0u);
            __asm__ volatile("global_load_async_to_lds_b128 %0, %1, off"
                             :: "v"(ldst), "v"(gfeat) : "memory");
            __asm__ volatile("global_load_async_to_lds_b128 %0, %1, off offset:16"
                             :: "v"(ldst), "v"(gfeat) : "memory");
            gfeat += (unsigned long long)K * 4;
        }

        // -- phase 3: y = p x E^T via 4 chained WMMAs -----------------------
        // A layout (16x32 f16): lane m<16 holds K=k0..k0+7 then k0+16..k0+23;
        // lane m+16 holds K=k0+8..k0+15 then k0+24..k0+31.
        v8f acc = {};
        const int aoff = half * 8;
#pragma unroll
        for (int kk = 0; kk < 4; ++kk) {
            const int k0 = kk * 32;
            v8h lo = *(const v8h*)&p_lds[l15 * PS + k0 + aoff];
            v8h hi = *(const v8h*)&p_lds[l15 * PS + k0 + 16 + aoff];
            v16h a = __builtin_shufflevector(lo, hi,
                0, 1, 2, 3, 4, 5, 6, 7, 8, 9, 10, 11, 12, 13, 14, 15);
            acc = __builtin_amdgcn_wmma_f32_16x16x32_f16(
                /*neg_a=*/false, a, /*neg_b=*/false, bf[kk],
                /*c_mod=*/(short)0, acc, /*reuse_a=*/false, /*reuse_b=*/false);
        }

        // -- phase 4: alpha' = log2(y)*ln2 + rowmax + feat[t] ---------------
        // D layout: VGPR r -> (M = half*8 + r, N = nb + l15).
        {
            const float* ft = &feat_lds[(t & 1) ? FBUF : 0];
            const int nn = nb + l15;
            const int mb = half * 8;
#pragma unroll
            for (int r = 0; r < 8; ++r) {
                const int mm = mb + r;
                const float y  = acc[r];
                const float fv = ft[mm * K + nn];
                alpha_lds[mm * AS + nn] =
                    __builtin_amdgcn_logf(y) * LN2 + rowmax[mm] + fv;
            }
        }
        __syncthreads();
    }

    // ---- terminal: forward_b = LSE_i(alpha[b][i] + trans[STOP][i]) --------
    if (tid < 16) {
        const float* ts = trans + CRF_STOP * K;
        float mm = -3.4e38f;
        for (int i = 0; i < K; ++i)
            mm = fmaxf(mm, alpha_lds[tid * AS + i] + ts[i]);
        float s = 0.0f;
        for (int i = 0; i < K; ++i)
            s += __builtin_amdgcn_exp2f(
                (alpha_lds[tid * AS + i] + ts[i] - mm) * LOG2E);
        atomicAdd(out, __builtin_amdgcn_logf(s) * LN2 + mm);
    }
}

__global__ __launch_bounds__(256) void crf_gold_kernel(
    const float* __restrict__ feats,   // [B][T][K]
    const int*   __restrict__ tags,    // [B][T]
    const float* __restrict__ trans,   // [K][K]
    float* __restrict__ out)
{
    const int b = blockIdx.x * blockDim.x + threadIdx.x;
    if (b >= CRF_B) return;
    constexpr int K = CRF_K, T = CRF_T;
    const int*   tb = tags  + (size_t)b * T;
    const float* fb = feats + (size_t)b * T * K;
    int   prev = CRF_START;
    float g    = 0.0f;
    for (int t = 0; t < T; ++t) {
        const int tg = tb[t];
        g += trans[tg * K + prev] + fb[(size_t)t * K + tg];
        prev = tg;
    }
    g += trans[CRF_STOP * K + prev];
    atomicAdd(out, -g);   // reference returns forward - gold
}

extern "C" void kernel_launch(void* const* d_in, const int* in_sizes, int n_in,
                              void* d_out, int out_size, void* d_ws, size_t ws_size,
                              hipStream_t stream) {
    (void)in_sizes; (void)n_in; (void)out_size; (void)d_ws; (void)ws_size;
    const float* feats = (const float*)d_in[0];
    const int*   tags  = (const int*)d_in[1];
    // d_in[2] = lengths: unused by the reference computation.
    const float* trans = (const float*)d_in[3];
    float* out = (float*)d_out;

    hipMemsetAsync(out, 0, sizeof(float), stream);
    crf_gold_kernel<<<2, 256, 0, stream>>>(feats, tags, trans, out);
    crf_forward_kernel<<<CRF_B / 16, 256, 0, stream>>>(feats, trans, out);
}
